// SwinTransformerBlock_51350628991023
// MI455X (gfx1250) — matrix operand order repaired and checked by reading, not compiled, not run on hardware
//
#include <hip/hip_runtime.h>
#include <hip/hip_bf16.h>
#include <math.h>
#include <stdint.h>

// ---------------- static config (Swin-T stage 1, SW-MSA) ----------------
#define B_      32
#define H_      56
#define W_      56
#define C_      96
#define NH_     3
#define WS_     7
#define SHIFT_  3
#define N_      49            // tokens per window
#define NW_     64            // windows per image
#define HD_     32            // head dim
#define M_      (B_ * NW_ * N_)   // 100352 rows (multiple of 128)
#define SCALE_  0.17677669529663687f
#define EPS_    1e-3f

typedef __attribute__((ext_vector_type(16))) _Float16 v16h;
typedef __attribute__((ext_vector_type(8)))  _Float16 v8h;
typedef __attribute__((ext_vector_type(8)))  float    v8f;
typedef __attribute__((ext_vector_type(4)))  unsigned int u32x4;
typedef __attribute__((ext_vector_type(8)))  int      i32x8;
typedef __attribute__((ext_vector_type(4)))  int      i32x4;

#if defined(__HIP_DEVICE_COMPILE__) && defined(__gfx1250__) && __has_builtin(__builtin_amdgcn_tensor_load_to_lds)
#define USE_TDM 1
#else
#define USE_TDM 0
#endif

// ---------------- fragment loaders (CDNA5 16-bit WMMA layouts) ----------------
// A 16x32 f16: lane l holds row M=l&15; a[i] = K = 16*(l>>4)+i (contiguous)
__device__ inline v16h ld_afrag(const _Float16* rowp, int kl) {
  v8h lo = *(const v8h*)(rowp + 2 * kl);
  v8h hi = *(const v8h*)(rowp + 2 * kl + 8);
  v16h r;
#pragma unroll
  for (int i = 0; i < 8; ++i) { r[i] = lo[i]; r[i + 8] = hi[i]; }
  return r;
}
// B 32x16 f16 (stored col-major in LDS): lane l holds col N=l&15; b[i] = K = 16*(l>>4)+i (contiguous)
__device__ inline v16h ld_bfrag(const _Float16* p) {
  v8h lo = *(const v8h*)p;
  v8h hi = *(const v8h*)(p + 8);
  v16h r;
#pragma unroll
  for (int i = 0; i < 8; ++i) { r[i] = lo[i]; r[i + 8] = hi[i]; }
  return r;
}

__device__ inline float red_sum32(float v) {
#pragma unroll
  for (int o = 1; o < 32; o <<= 1) v += __shfl_xor(v, o, 32);
  return v;
}
__device__ inline float red_max16(float v) {
#pragma unroll
  for (int o = 1; o < 16; o <<= 1) v = fmaxf(v, __shfl_xor(v, o, 32));
  return v;
}
__device__ inline float red_sum16(float v) {
#pragma unroll
  for (int o = 1; o < 16; o <<= 1) v += __shfl_xor(v, o, 32);
  return v;
}
__device__ inline float gelu_exact(float x) {
  return 0.5f * x * (1.0f + erff(x * 0.70710678118654752f));
}

// ---------------- TDM: async 2D tile load Global -> LDS (Tensor Data Mover) ----------------
#if USE_TDM
// Loads tile_y rows of tile_x f16 elements (row stride = stride_elems in memory) into LDS at
// lds_addr, inserting 4 DWORDs (8 halves) of padding after every 16 DWORDs (32 halves):
// matches an LDS row stride of 40 halves for tile_x == 32.
__device__ inline void tdm_load_a_tile(unsigned lds_addr, const _Float16* gptr,
                                       unsigned tile_x, unsigned tile_y, unsigned stride_elems) {
  unsigned long long ga = (unsigned long long)(size_t)gptr;
  u32x4 g0;
  g0[0] = 1u;                                            // count=1, user descriptor
  g0[1] = lds_addr;                                      // LDS byte address
  g0[2] = (unsigned)(ga & 0xFFFFFFFFu);                  // global addr [31:0]
  g0[3] = (unsigned)((ga >> 32) & 0x1FFFFFFu) | (2u << 30);  // addr[56:32], type=2 (image)
  i32x8 g1;
  g1[0] = (1 << 16)       // data_size = 2 bytes
        | (1 << 20)       // pad_enable
        | (3 << 22)       // pad_interval: 16 DWORDs of data
        | (3 << 25);      // pad_amount: 4 DWORDs of pad
  g1[1] = (int)((tile_x & 0xFFFFu) << 16);               // tensor_dim0[15:0] (bits 63:48)
  g1[2] = (int)(((tile_x >> 16) & 0xFFFFu) | ((tile_y & 0xFFFFu) << 16)); // dim0 hi / dim1 lo
  g1[3] = (int)(((tile_y >> 16) & 0xFFFFu) | ((tile_x & 0xFFFFu) << 16)); // dim1 hi / tile_dim0
  g1[4] = (int)(tile_y & 0xFFFFu);                       // tile_dim1 (tile_dim2 = 0)
  g1[5] = (int)stride_elems;                             // tensor_dim0_stride [31:0]
  g1[6] = 0;                                             // stride hi / dim1_stride lo
  g1[7] = 0;
  i32x4 z4 = {0, 0, 0, 0};
#if __clang_major__ >= 23
  i32x8 z8 = {0, 0, 0, 0, 0, 0, 0, 0};
  __builtin_amdgcn_tensor_load_to_lds(g0, g1, z4, z4, z8, 0);
#else
  __builtin_amdgcn_tensor_load_to_lds(g0, g1, z4, z4, 0);
#endif
}
#endif

// ---------------- kernel: f32 -> f16 convert ----------------
__global__ void cvt_f16_kernel(const float* __restrict__ src, _Float16* __restrict__ dst, int n) {
  int i = blockIdx.x * 256 + threadIdx.x;
  if (i < n) dst[i] = (_Float16)src[i];
}

// ---------------- kernel: fused (rel-pos bias + shift mask) LUT ----------------
// lut[wloc][head][i][j], i,j in [0,64): bias+mask for valid (i<49,j<49), else -1e30
__device__ inline int region_group(int wy, int wx, int n) {
  int y = wy * WS_ + n / WS_;
  int x = wx * WS_ + n % WS_;
  int ry = (y < H_ - WS_) ? 0 : ((y < H_ - SHIFT_) ? 1 : 2);
  int rx = (x < W_ - WS_) ? 0 : ((x < W_ - SHIFT_) ? 1 : 2);
  return ry * 3 + rx;
}

__global__ __launch_bounds__(256) void build_lut_kernel(
    const float* __restrict__ bias_table, float* __restrict__ lut) {
  int head = blockIdx.x % NH_;
  int wloc = blockIdx.x / NH_;
  int wy = wloc / (W_ / WS_), wx = wloc % (W_ / WS_);
  float* dst = lut + (size_t)blockIdx.x * 4096;
  for (int e = threadIdx.x; e < 4096; e += 256) {
    int i = e >> 6, j = e & 63;
    float v = -1e30f;
    if (i < N_ && j < N_) {
      int ai = i / WS_, bi = i % WS_, aj = j / WS_, bj = j % WS_;
      int rel = ((bi - bj) + WS_ - 1) * (2 * WS_ - 1) + (ai - aj) + WS_ - 1;
      v = bias_table[rel * NH_ + head];
      if (region_group(wy, wx, i) != region_group(wy, wx, j)) v += -100.0f;
    }
    dst[e] = v;
  }
}

// ---------------- kernel: LN1 + cyclic shift + window partition ----------------
__global__ __launch_bounds__(256) void ln1_shift_kernel(
    const float* __restrict__ x, const float* __restrict__ g, const float* __restrict__ b,
    _Float16* __restrict__ h16) {
  int wave = threadIdx.x >> 5, lane = threadIdx.x & 31;
  int t = blockIdx.x * 8 + wave;
  int bb  = t / (NW_ * N_);
  int rem = t - bb * (NW_ * N_);
  int win = rem / N_;
  int n   = rem - win * N_;
  int wy = win / (W_ / WS_), wx = win % (W_ / WS_);
  int y  = wy * WS_ + n / WS_, xx = wx * WS_ + n % WS_;
  int ys = (y + SHIFT_) % H_, xs = (xx + SHIFT_) % W_;   // roll(-SHIFT): h[y]=x[(y+SHIFT)%H]
  const float* src = x + ((size_t)bb * (H_ * W_) + (size_t)ys * W_ + xs) * C_;
  float v[3];
#pragma unroll
  for (int r = 0; r < 3; ++r) v[r] = src[lane + 32 * r];
  float mu = red_sum32(v[0] + v[1] + v[2]) * (1.0f / C_);
  float d0 = v[0] - mu, d1 = v[1] - mu, d2 = v[2] - mu;
  float var = red_sum32(d0 * d0 + d1 * d1 + d2 * d2) * (1.0f / C_);
  float inv = rsqrtf(var + EPS_);
#pragma unroll
  for (int r = 0; r < 3; ++r) {
    int c = lane + 32 * r;
    h16[(size_t)t * C_ + c] = (_Float16)((v[r] - mu) * inv * g[c] + b[c]);
  }
}

// ---------------- kernel: LN2 (identity token order) ----------------
__global__ __launch_bounds__(256) void ln2_kernel(
    const float* __restrict__ x2, const float* __restrict__ g, const float* __restrict__ b,
    _Float16* __restrict__ y16) {
  int wave = threadIdx.x >> 5, lane = threadIdx.x & 31;
  int t = blockIdx.x * 8 + wave;
  const float* src = x2 + (size_t)t * C_;
  float v[3];
#pragma unroll
  for (int r = 0; r < 3; ++r) v[r] = src[lane + 32 * r];
  float mu = red_sum32(v[0] + v[1] + v[2]) * (1.0f / C_);
  float d0 = v[0] - mu, d1 = v[1] - mu, d2 = v[2] - mu;
  float var = red_sum32(d0 * d0 + d1 * d1 + d2 * d2) * (1.0f / C_);
  float inv = rsqrtf(var + EPS_);
#pragma unroll
  for (int r = 0; r < 3; ++r) {
    int c = lane + 32 * r;
    y16[(size_t)t * C_ + c] = (_Float16)((v[r] - mu) * inv * g[c] + b[c]);
  }
}

// ---------------- WMMA GEMM: C[M,N] = A[M,K](f16) * B[K,N](f16) + bias, fused epilogues --------
enum { EPI_QKV = 0, EPI_PROJ = 1, EPI_FC1 = 2, EPI_FC2 = 3 };

template <int EPI>
__global__ __launch_bounds__(256) void gemm_wmma_kernel(
    const _Float16* __restrict__ A, const _Float16* __restrict__ Bw,
    const float* __restrict__ bias, int K, int Nfull,
    _Float16* __restrict__ out16, float* __restrict__ out32,
    const float* __restrict__ resid) {
  __shared__ _Float16 As[128 * 40];   // [row][k] stride 40 halves (80B, 16B aligned)
  __shared__ _Float16 Bt[96 * 40];    // transposed: [col][k] stride 40 halves

  int tid = threadIdx.x;
  int wave = tid >> 5, lane = tid & 31;
  int rowBase = blockIdx.x * 128;
  int colBase = blockIdx.y * 96;

  v8f acc[6];
#pragma unroll
  for (int i = 0; i < 6; ++i)
#pragma unroll
    for (int j = 0; j < 8; ++j) acc[i][j] = 0.0f;

  int m  = lane & 15;
  int kl = (lane >> 4) * 8;
  int kb = (lane >> 4) * 16;

  for (int k0 = 0; k0 < K; k0 += 32) {
    // ---- stage A tile 128x32 into LDS ----
#if USE_TDM
    if (wave == 0) {   // one TDM DMA per block (EXEC-independent, issued per-wave)
      tdm_load_a_tile((unsigned)(size_t)&As[0],
                      A + (size_t)rowBase * K + k0, 32u, 128u, (unsigned)K);
      __builtin_amdgcn_s_wait_tensorcnt(0);
    }
#else
    {
      int r = tid >> 1;
      int c = (tid & 1) * 16;
      const uint4* s4 = (const uint4*)(A + (size_t)(rowBase + r) * K + k0 + c);
      uint4* d4 = (uint4*)&As[r * 40 + c];
      d4[0] = s4[0];
      d4[1] = s4[1];
    }
#endif
    // ---- stage B tile 32x96 transposed into Bt[col][k] ----
    if (tid < 192) {
      int r  = tid / 6;          // k within tile 0..31
      int c0 = (tid % 6) * 16;   // col start
      const _Float16* src = Bw + (size_t)(k0 + r) * Nfull + colBase + c0;
#pragma unroll
      for (int i = 0; i < 16; ++i) Bt[(c0 + i) * 40 + r] = src[i];
    }
    __syncthreads();

    v16h afrag = ld_afrag(&As[(wave * 16 + m) * 40], kl);
#pragma unroll
    for (int nt = 0; nt < 6; ++nt) {
      v16h bfrag = ld_bfrag(&Bt[(nt * 16 + m) * 40 + kb]);
      acc[nt] = __builtin_amdgcn_wmma_f32_16x16x32_f16(
          false, afrag, false, bfrag, (short)0, acc[nt], false, false);
    }
    __syncthreads();
  }

  // epilogue: lane holds (row = 8*(lane>>4)+j, col = lane&15) of each 16x16 tile
  int rgrp = (lane >> 4) * 8;
#pragma unroll
  for (int nt = 0; nt < 6; ++nt) {
    int col = colBase + nt * 16 + (lane & 15);
    float bv = bias[col];
#pragma unroll
    for (int j = 0; j < 8; ++j) {
      int row = rowBase + wave * 16 + rgrp + j;
      float v = acc[nt][j] + bv;
      if (EPI == EPI_QKV) {
        out16[(size_t)row * Nfull + col] = (_Float16)v;
      } else if (EPI == EPI_FC1) {
        out16[(size_t)row * Nfull + col] = (_Float16)gelu_exact(gelu_exact(v));
      } else if (EPI == EPI_FC2) {
        size_t di = (size_t)row * Nfull + col;
        out32[di] = v + resid[di];
      } else {  // EPI_PROJ: window-reverse + reverse shift scatter, residual add of input x
        int t   = row;
        int bb  = t / (NW_ * N_);
        int rem = t - bb * (NW_ * N_);
        int win = rem / N_;
        int n   = rem - win * N_;
        int wy = win / (W_ / WS_), wx = win % (W_ / WS_);
        int y  = wy * WS_ + n / WS_, xx = wx * WS_ + n % WS_;
        int yd = (y + SHIFT_) % H_, xd = (xx + SHIFT_) % W_;
        size_t di = ((size_t)bb * (H_ * W_) + (size_t)yd * W_ + xd) * C_ + col;
        out32[di] = resid[di] + v;
      }
    }
  }
}

// ---------------- kernel: windowed attention (one block per (window, head)) ----------------
__global__ __launch_bounds__(128) void attn_kernel(
    const _Float16* __restrict__ qkv, const float* __restrict__ lut,
    _Float16* __restrict__ o16) {
  __shared__ _Float16 q16[64 * 40];   // [query][dim], q pre-scaled
  __shared__ _Float16 k16[64 * 40];   // [key][dim]
  __shared__ _Float16 vT[32 * 72];    // [dim][key] (transposed V)
  __shared__ _Float16 p16[64 * 72];   // softmaxed probs [query][key]

  int bid    = blockIdx.x;
  int head   = bid % NH_;
  int winIdx = bid / NH_;             // b*NW + local window
  int wloc   = winIdx % NW_;
  int tid = threadIdx.x, wave = tid >> 5, lane = tid & 31;

  const _Float16* base = qkv + (size_t)winIdx * N_ * (3 * C_);
  const float* lutp = lut + ((size_t)wloc * NH_ + head) * 4096;

  // stage q (scaled), k, v^T; zero padding rows/cols 49..63
  for (int idx = tid; idx < 64 * 32; idx += 128) {
    int n = idx >> 5, d = idx & 31;
    _Float16 qv = (_Float16)0.0f, kv = (_Float16)0.0f, vv = (_Float16)0.0f;
    if (n < N_) {
      const _Float16* p = base + (size_t)n * (3 * C_) + head * HD_ + d;
      qv = (_Float16)((float)p[0] * SCALE_);
      kv = p[C_];
      vv = p[2 * C_];
    }
    q16[n * 40 + d] = qv;
    k16[n * 40 + d] = kv;
    vT[d * 72 + n]  = vv;
  }
  __syncthreads();

  // S = q * k^T  (each wave: m-tile = wave, 4 n-tiles, K = 32)
  v8f s[4];
#pragma unroll
  for (int i = 0; i < 4; ++i)
#pragma unroll
    for (int j = 0; j < 8; ++j) s[i][j] = 0.0f;

  int cl = lane & 15;
  int kl = (lane >> 4) * 8;
  int kb = (lane >> 4) * 16;
  {
    v16h aq = ld_afrag(&q16[(wave * 16 + cl) * 40], kl);
#pragma unroll
    for (int nt = 0; nt < 4; ++nt) {
      v16h bk = ld_bfrag(&k16[(nt * 16 + cl) * 40 + kb]);
      s[nt] = __builtin_amdgcn_wmma_f32_16x16x32_f16(
          false, aq, false, bk, (short)0, s[nt], false, false);
    }
  }

  // + fused (rel-pos bias + shift mask) from LUT; invalid entries get -1e30
  int rbase = wave * 16 + (lane >> 4) * 8;
#pragma unroll
  for (int nt = 0; nt < 4; ++nt) {
    int col = nt * 16 + cl;
#pragma unroll
    for (int j = 0; j < 8; ++j) s[nt][j] += lutp[(rbase + j) * 64 + col];
  }

  // double softmax over valid 49 cols (reference applies softmax twice)
#pragma unroll
  for (int pass = 0; pass < 2; ++pass) {
#pragma unroll
    for (int j = 0; j < 8; ++j) {
      int row = rbase + j;
      float mx = -1e30f;
#pragma unroll
      for (int nt = 0; nt < 4; ++nt) {
        int col = nt * 16 + cl;
        float v = (row < N_ && col < N_) ? s[nt][j] : -1e30f;
        mx = fmaxf(mx, v);
      }
      mx = red_max16(mx);
      float sum = 0.0f;
#pragma unroll
      for (int nt = 0; nt < 4; ++nt) {
        int col = nt * 16 + cl;
        float e = (row < N_ && col < N_) ? __expf(s[nt][j] - mx) : 0.0f;
        s[nt][j] = e;
        sum += e;
      }
      sum = red_sum16(sum);
      float inv = 1.0f / sum;
#pragma unroll
      for (int nt = 0; nt < 4; ++nt) s[nt][j] *= inv;
    }
  }

  // write P (f16) into LDS
#pragma unroll
  for (int nt = 0; nt < 4; ++nt) {
    int col = nt * 16 + cl;
#pragma unroll
    for (int j = 0; j < 8; ++j) p16[(rbase + j) * 72 + col] = (_Float16)s[nt][j];
  }
  __syncthreads();

  // O = P * V  (K = 64 -> 2 k-steps, N = 32 -> 2 n-tiles)
  v8f o[2];
#pragma unroll
  for (int i = 0; i < 2; ++i)
#pragma unroll
    for (int j = 0; j < 8; ++j) o[i][j] = 0.0f;
#pragma unroll
  for (int kk = 0; kk < 2; ++kk) {
    v16h ap = ld_afrag(&p16[(wave * 16 + cl) * 72 + kk * 32], kl);
#pragma unroll
    for (int nt = 0; nt < 2; ++nt) {
      v16h bv = ld_bfrag(&vT[(nt * 16 + cl) * 72 + kk * 32 + kb]);
      o[nt] = __builtin_amdgcn_wmma_f32_16x16x32_f16(
          false, ap, false, bv, (short)0, o[nt], false, false);
    }
  }

#pragma unroll
  for (int nt = 0; nt < 2; ++nt) {
    int d = nt * 16 + cl;
#pragma unroll
    for (int j = 0; j < 8; ++j) {
      int r = rbase + j;
      if (r < N_)
        o16[((size_t)winIdx * N_ + r) * C_ + head * HD_ + d] = (_Float16)o[nt][j];
    }
  }
}

// ---------------- launcher ----------------
extern "C" void kernel_launch(void* const* d_in, const int* in_sizes, int n_in,
                              void* d_out, int out_size, void* d_ws, size_t ws_size,
                              hipStream_t stream) {
  const float* x          = (const float*)d_in[0];
  const float* norm1_g    = (const float*)d_in[1];
  const float* norm1_b    = (const float*)d_in[2];
  const float* qkv_w      = (const float*)d_in[3];
  const float* qkv_b      = (const float*)d_in[4];
  const float* bias_table = (const float*)d_in[5];
  const float* proj_w     = (const float*)d_in[6];
  const float* proj_b     = (const float*)d_in[7];
  const float* norm2_g    = (const float*)d_in[8];
  const float* norm2_b    = (const float*)d_in[9];
  const float* fc1_w      = (const float*)d_in[10];
  const float* fc1_b      = (const float*)d_in[11];
  const float* fc2_w      = (const float*)d_in[12];
  const float* fc2_b      = (const float*)d_in[13];
  float* out = (float*)d_out;

  // workspace carve-up (256B aligned)
  char* ws = (char*)d_ws;
  size_t off = 0;
  auto take = [&](size_t bytes) -> char* {
    char* p = ws + off;
    off = (off + bytes + 255) & ~(size_t)255;
    return p;
  };
  _Float16* wqkv16  = (_Float16*)take((size_t)C_ * 3 * C_ * 2);
  _Float16* wproj16 = (_Float16*)take((size_t)C_ * C_ * 2);
  _Float16* wfc1_16 = (_Float16*)take((size_t)C_ * 4 * C_ * 2);
  _Float16* wfc2_16 = (_Float16*)take((size_t)4 * C_ * C_ * 2);
  float*    lut     = (float*)take((size_t)NW_ * NH_ * 64 * 64 * 4);
  _Float16* h16   = (_Float16*)take((size_t)M_ * C_ * 2);
  _Float16* qkv16 = (_Float16*)take((size_t)M_ * 3 * C_ * 2);
  float*    x2    = (float*)take((size_t)M_ * C_ * 4);
  _Float16* m16   = (_Float16*)take((size_t)M_ * 4 * C_ * 2);
  _Float16* o16 = h16;     // h16 dead after QKV GEMM
  _Float16* y16 = qkv16;   // qkv16 dead after attention

  // 1) weights -> f16; fused bias+mask LUT
  auto cvt = [&](const float* s, _Float16* d, int n) {
    cvt_f16_kernel<<<(n + 255) / 256, 256, 0, stream>>>(s, d, n);
  };
  cvt(qkv_w, wqkv16, C_ * 3 * C_);
  cvt(proj_w, wproj16, C_ * C_);
  cvt(fc1_w, wfc1_16, C_ * 4 * C_);
  cvt(fc2_w, wfc2_16, 4 * C_ * C_);
  build_lut_kernel<<<NW_ * NH_, 256, 0, stream>>>(bias_table, lut);

  // 2) LN1 + shift + window partition
  ln1_shift_kernel<<<M_ / 8, 256, 0, stream>>>(x, norm1_g, norm1_b, h16);

  // 3) QKV GEMM: [M,96] x [96,288]
  gemm_wmma_kernel<EPI_QKV><<<dim3(M_ / 128, 3), 256, 0, stream>>>(
      h16, wqkv16, qkv_b, C_, 3 * C_, qkv16, nullptr, nullptr);

  // 4) windowed attention (B*NW windows x NH heads)
  attn_kernel<<<B_ * NW_ * NH_, 128, 0, stream>>>(qkv16, lut, o16);

  // 5) proj GEMM + window-reverse + reverse shift + residual -> x2 (f32)
  gemm_wmma_kernel<EPI_PROJ><<<dim3(M_ / 128, 1), 256, 0, stream>>>(
      o16, wproj16, proj_b, C_, C_, nullptr, x2, x);

  // 6) LN2
  ln2_kernel<<<M_ / 8, 256, 0, stream>>>(x2, norm2_g, norm2_b, y16);

  // 7) fc1 GEMM + bias + double exact GELU
  gemm_wmma_kernel<EPI_FC1><<<dim3(M_ / 128, 4), 256, 0, stream>>>(
      y16, wfc1_16, fc1_b, C_, 4 * C_, m16, nullptr, nullptr);

  // 8) fc2 GEMM + bias + residual -> out
  gemm_wmma_kernel<EPI_FC2><<<dim3(M_ / 128, 1), 256, 0, stream>>>(
      m16, wfc2_16, fc2_b, 4 * C_, C_, nullptr, out, x2);
}
